// RNNLayer_40450001993750
// MI455X (gfx1250) — matrix operand order repaired and checked by reading, not compiled, not run on hardware
//
#include <hip/hip_runtime.h>
#include <cmath>

// ---------------------------------------------------------------------------
// Types for CDNA5 WMMA (gfx1250, wave32)
// ---------------------------------------------------------------------------
typedef __attribute__((ext_vector_type(16))) __bf16 v16bf;
typedef __attribute__((ext_vector_type(8)))  float  v8f;
typedef __attribute__((ext_vector_type(4)))  float  v4f;
typedef __attribute__((ext_vector_type(4)))  unsigned int   v4u;
typedef __attribute__((ext_vector_type(4)))  unsigned short v4s;

union Frag16 {
    v16bf          v;     // WMMA operand (16 bf16 = 8 VGPRs)
    unsigned short u[16];
    v4u            q[2];  // two 128-bit chunks for loads
};

#define HDIM   1024
#define TSTEPS 512
#define BATCH  64

// bf16 round-to-nearest-even from fp32
__device__ __forceinline__ unsigned short f2bf_rne(float f) {
    unsigned u = __float_as_uint(f);
    u += 0x7FFFu + ((u >> 16) & 1u);
    return (unsigned short)(u >> 16);
}
__device__ __forceinline__ float bf2f(unsigned short h) {
    return __uint_as_float(((unsigned)h) << 16);
}

// Load 8 consecutive fp32, split into bf16 hi/lo at elements [e0..e0+7]
// (fallback path only)
__device__ __forceinline__ void split8(const float* __restrict__ p,
                                       Frag16& hi, Frag16& lo, int e0) {
    v4f a = *(const v4f*)p;
    v4f b = *(const v4f*)(p + 4);
    float f[8] = {a.x, a.y, a.z, a.w, b.x, b.y, b.z, b.w};
#pragma unroll
    for (int i = 0; i < 8; ++i) {
        unsigned short h = f2bf_rne(f[i]);
        hi.u[e0 + i] = h;
        lo.u[e0 + i] = f2bf_rne(f[i] - bf2f(h));
    }
}

// B fragment slice: elements 0..7 at p, elements 8..15 at p+16 (two 128-bit loads)
__device__ __forceinline__ void load_b16(const unsigned short* __restrict__ p,
                                         Frag16& fr) {
    fr.q[0] = *(const v4u*)p;
    fr.q[1] = *(const v4u*)(p + 16);
}

// A fragment from pre-split bf16: elements 0..7 at p, elements 8..15 at p+16
// (K-base and K-base+16 per the ISA 16-bit A 16x32 layout)
__device__ __forceinline__ void load_a16(const unsigned short* __restrict__ p,
                                         Frag16& fr) {
    fr.q[0] = *(const v4u*)p;
    fr.q[1] = *(const v4u*)(p + 16);
}

__device__ __forceinline__ v8f wmma_bf16(const Frag16& a, const Frag16& b, v8f c) {
    return __builtin_amdgcn_wmma_f32_16x16x32_bf16(
        false, a.v, false, b.v, (short)0, c, false, false);
}

// ---------------------------------------------------------------------------
// Kernel 0: generic fp32 -> (bf16 hi, bf16 lo) split, 4 elements/thread
// ---------------------------------------------------------------------------
__global__ __launch_bounds__(256) void split_kernel(
    const float* __restrict__ src,
    unsigned short* __restrict__ hi, unsigned short* __restrict__ lo, int n4) {
    int i = blockIdx.x * 256 + threadIdx.x;
    if (i >= n4) return;
    v4f f = *(const v4f*)(src + (size_t)i * 4);
    v4s h, l;
#pragma unroll
    for (int j = 0; j < 4; ++j) {
        unsigned short hb = f2bf_rne(f[j]);
        h[j] = hb;
        l[j] = f2bf_rne(f[j] - bf2f(hb));
    }
    *(v4s*)(hi + (size_t)i * 4) = h;
    *(v4s*)(lo + (size_t)i * 4) = l;
}

// ---------------------------------------------------------------------------
// Kernel 1a (fast path): x_proj from PRE-SPLIT bf16 input.
// M=32768, N=K=1024. Block = 8 waves, tile 128(M)x64(N).
// Wave: 2 M-tiles x 2 N-tiles -> 12 WMMA : 16 b128 loads per k-step, zero
// conversion VALU in the loop.
// ---------------------------------------------------------------------------
__global__ __launch_bounds__(256) void xproj_presplit_kernel(
    const unsigned short* __restrict__ Ah,       // input hi [32768,1024] bf16
    const unsigned short* __restrict__ Al,       // input lo
    const unsigned short* __restrict__ Wh,       // W_ih hi  [1024,1024] bf16
    const unsigned short* __restrict__ Wl,       // W_ih lo
    const float* __restrict__ b_ih, const float* __restrict__ b_hh,
    float* __restrict__ C)                       // [32768,1024] fp32
{
    const int lane = threadIdx.x & 31;
    const int w    = threadIdx.x >> 5;
    const int lr   = lane & 15;
    const int half = lane >> 4;
    const int wm   = w & 3;
    const int wn   = w >> 2;
    const int m0   = blockIdx.y * 128 + wm * 32;
    const int n0   = blockIdx.x * 64 + wn * 32;

    v8f acc[2][2] = {{v8f{}, v8f{}}, {v8f{}, v8f{}}};

    for (int k0 = 0; k0 < HDIM; k0 += 32) {
        Frag16 ah[2], al[2], bh[2], bl[2];
        const int ka = k0 + half * 8;    // A lane K-base
        const int kb = k0 + half * 8;    // B lane K-base
#pragma unroll
        for (int mi = 0; mi < 2; ++mi) {
            const size_t ar = (size_t)(m0 + mi * 16 + lr) * HDIM + ka;
            load_a16(Ah + ar, ah[mi]);
            load_a16(Al + ar, al[mi]);
        }
#pragma unroll
        for (int ni = 0; ni < 2; ++ni) {
            const size_t br = (size_t)(n0 + ni * 16 + lr) * HDIM + kb;
            load_b16(Wh + br, bh[ni]);
            load_b16(Wl + br, bl[ni]);
        }
#pragma unroll
        for (int mi = 0; mi < 2; ++mi)
#pragma unroll
            for (int ni = 0; ni < 2; ++ni) {
                acc[mi][ni] = wmma_bf16(ah[mi], bh[ni], acc[mi][ni]);
                acc[mi][ni] = wmma_bf16(ah[mi], bl[ni], acc[mi][ni]);
                acc[mi][ni] = wmma_bf16(al[mi], bh[ni], acc[mi][ni]);
            }
    }

#pragma unroll
    for (int mi = 0; mi < 2; ++mi) {
        const int mbase = m0 + mi * 16 + half * 8;
#pragma unroll
        for (int ni = 0; ni < 2; ++ni) {
            const int n = n0 + ni * 16 + lr;
            const float bias = b_ih[n] + b_hh[n];
#pragma unroll
            for (int r = 0; r < 8; ++r)
                C[(size_t)(mbase + r) * HDIM + n] = acc[mi][ni][r] + bias;
        }
    }
}

// ---------------------------------------------------------------------------
// Kernel 1b (fallback if ws too small): on-the-fly split of fp32 input.
// ---------------------------------------------------------------------------
__global__ __launch_bounds__(256) void xproj_otf_kernel(
    const float* __restrict__ A,
    const unsigned short* __restrict__ Wh, const unsigned short* __restrict__ Wl,
    const float* __restrict__ b_ih, const float* __restrict__ b_hh,
    float* __restrict__ C)
{
    const int lane = threadIdx.x & 31;
    const int w    = threadIdx.x >> 5;
    const int lr   = lane & 15;
    const int half = lane >> 4;
    const int m0   = blockIdx.y * 128 + w * 16;
    const int n0b  = blockIdx.x * 64;
    const size_t arow = (size_t)(m0 + lr);

    v8f acc[4] = {v8f{}, v8f{}, v8f{}, v8f{}};

    for (int k0 = 0; k0 < HDIM; k0 += 32) {
        Frag16 ah, al;
        const float* ap = A + arow * HDIM + (k0 + half * 8);
        split8(ap,      ah, al, 0);
        split8(ap + 16, ah, al, 8);
        const int kb = k0 + half * 8;
#pragma unroll
        for (int j = 0; j < 4; ++j) {
            Frag16 bh, bl;
            const size_t brow = (size_t)(n0b + j * 16 + lr);
            load_b16(Wh + brow * HDIM + kb, bh);
            load_b16(Wl + brow * HDIM + kb, bl);
            acc[j] = wmma_bf16(ah, bh, acc[j]);
            acc[j] = wmma_bf16(ah, bl, acc[j]);
            acc[j] = wmma_bf16(al, bh, acc[j]);
        }
    }

    const int mbase = m0 + half * 8;
#pragma unroll
    for (int j = 0; j < 4; ++j) {
        const int n = n0b + j * 16 + lr;
        const float bias = b_ih[n] + b_hh[n];
#pragma unroll
        for (int r = 0; r < 8; ++r)
            C[(size_t)(mbase + r) * HDIM + n] = acc[j][r] + bias;
    }
}

// ---------------------------------------------------------------------------
// Kernel 2: one recurrence step, reading PRE-SPLIT h and writing the split of
// the new h for the next step (double-buffered to avoid cross-block races).
//   xp[b,n] = tanh(xp[b,n] + sum_k h[b,k]*W_hh[n,k])
// Grid (32,4): block = 16(M) x 32(N); 8 waves = 4 K-quarters x 2 N-tiles,
// partials reduced through LDS. Inner loop: 8 b128 loads + 3 WMMA, no VALU.
// ---------------------------------------------------------------------------
__global__ __launch_bounds__(256) void rnn_step_kernel(
    const unsigned short* __restrict__ Hh,   // h_{t-1} hi [64,1024] bf16
    const unsigned short* __restrict__ Hl,   // h_{t-1} lo
    float* __restrict__ xp,                  // [64,1024] fp32 (in/out -> h_t)
    const unsigned short* __restrict__ Wh,   // W_hh hi
    const unsigned short* __restrict__ Wl,   // W_hh lo
    unsigned short* __restrict__ Oh,         // h_t hi (next step's input)
    unsigned short* __restrict__ Ol)         // h_t lo
{
    __shared__ float lds[8][256];

    const int lane = threadIdx.x & 31;
    const int w    = threadIdx.x >> 5;
    const int lr   = lane & 15;
    const int half = lane >> 4;
    const int kq   = w & 3;
    const int nt   = w >> 2;
    const int m0   = blockIdx.y * 16;
    const int nb   = blockIdx.x * 32;
    const int n0   = nb + nt * 16;

    v8f acc = v8f{};
    const int kbeg = kq * 256, kend = kbeg + 256;
    const size_t arow = (size_t)(m0 + lr) * HDIM;
    const size_t brow = (size_t)(n0 + lr) * HDIM;
    for (int k0 = kbeg; k0 < kend; k0 += 32) {
        Frag16 ah, al, bh, bl;
        load_a16(Hh + arow + k0 + half * 8, ah);
        load_a16(Hl + arow + k0 + half * 8, al);
        load_b16(Wh + brow + k0 + half * 8, bh);
        load_b16(Wl + brow + k0 + half * 8, bl);
        acc = wmma_bf16(ah, bh, acc);
        acc = wmma_bf16(ah, bl, acc);
        acc = wmma_bf16(al, bh, acc);
    }

#pragma unroll
    for (int r = 0; r < 8; ++r) lds[w][lane * 8 + r] = acc[r];
    __syncthreads();

    const int tid = threadIdx.x;
#pragma unroll
    for (int o = 0; o < 2; ++o) {
        const int f   = tid + o * 256;
        const int ntp = f >> 8;
        const int e   = f & 255;
        float s = lds[ntp * 4 + 0][e] + lds[ntp * 4 + 1][e] +
                  lds[ntp * 4 + 2][e] + lds[ntp * 4 + 3][e];
        const int lane2 = e >> 3, r2 = e & 7;
        const int m = m0 + ((lane2 >> 4) << 3) + r2;
        const int n = nb + ntp * 16 + (lane2 & 15);
        const size_t idx = (size_t)m * HDIM + n;
        const float val = tanhf(s + xp[idx]);
        xp[idx] = val;
        const unsigned short hb = f2bf_rne(val);
        Oh[idx] = hb;
        Ol[idx] = f2bf_rne(val - bf2f(hb));
    }
}

// ---------------------------------------------------------------------------
// Host-side launch
// ---------------------------------------------------------------------------
extern "C" void kernel_launch(void* const* d_in, const int* in_sizes, int n_in,
                              void* d_out, int out_size, void* d_ws, size_t ws_size,
                              hipStream_t stream) {
    (void)in_sizes; (void)n_in; (void)out_size;
    const float* input = (const float*)d_in[0];   // [T,B,IN]
    const float* h0    = (const float*)d_in[1];   // [1,B,H]
    const float* W_ih  = (const float*)d_in[2];   // [H,IN]
    const float* W_hh  = (const float*)d_in[3];   // [H,H]
    const float* b_ih  = (const float*)d_in[4];   // [H]
    const float* b_hh  = (const float*)d_in[5];   // [H]
    float* out = (float*)d_out;                   // [T,B,H] outputs + [B,H] h_last

    const size_t NW  = (size_t)HDIM * HDIM;       // 1M
    const size_t NH  = (size_t)BATCH * HDIM;      // 64K
    const size_t NX  = (size_t)TSTEPS * NH;       // 32M

    // ws layout (shorts): weights (4*NW), h double-buffer (4*NH), input split
    unsigned short* wih_hi = (unsigned short*)d_ws;
    unsigned short* wih_lo = wih_hi + NW;
    unsigned short* whh_hi = wih_lo + NW;
    unsigned short* whh_lo = whh_hi + NW;
    unsigned short* hs[2][2] = {
        {whh_lo + NW,          whh_lo + NW + NH},      // buf0: hi, lo
        {whh_lo + NW + 2 * NH, whh_lo + NW + 3 * NH},  // buf1: hi, lo
    };
    unsigned short* ain_hi = whh_lo + NW + 4 * NH;
    unsigned short* ain_lo = ain_hi + NX;
    const size_t need_fast = (size_t)((char*)(ain_lo + NX) - (char*)d_ws);
    const bool fast = ws_size >= need_fast;       // ~143 MB for fast path

    // 0) one-time bf16 hi/lo splits (bandwidth-bound; ~256MB traffic = ~11us)
    split_kernel<<<(NW / 4 + 255) / 256, 256, 0, stream>>>(W_ih, wih_hi, wih_lo, (int)(NW / 4));
    split_kernel<<<(NW / 4 + 255) / 256, 256, 0, stream>>>(W_hh, whh_hi, whh_lo, (int)(NW / 4));
    split_kernel<<<(NH / 4 + 255) / 256, 256, 0, stream>>>(h0, hs[0][0], hs[0][1], (int)(NH / 4));

    // 1) x_proj for all timesteps, staged directly in d_out (same shape)
    dim3 g1(HDIM / 64, (TSTEPS * BATCH) / 128);
    if (fast) {
        split_kernel<<<(NX / 4 + 255) / 256, 256, 0, stream>>>(input, ain_hi, ain_lo, (int)(NX / 4));
        xproj_presplit_kernel<<<g1, 256, 0, stream>>>(ain_hi, ain_lo, wih_hi, wih_lo,
                                                      b_ih, b_hh, out);
    } else {
        xproj_otf_kernel<<<g1, 256, 0, stream>>>(input, wih_hi, wih_lo, b_ih, b_hh, out);
    }

    // 2) sequential recurrence: step t reads h split buf[t&1], writes buf[~t&1]
    dim3 g2(HDIM / 32, BATCH / 16);
    for (int t = 0; t < TSTEPS; ++t) {
        const int bi = t & 1, bo = bi ^ 1;
        rnn_step_kernel<<<g2, 256, 0, stream>>>(
            hs[bi][0], hs[bi][1], out + (size_t)t * NH,
            whh_hi, whh_lo, hs[bo][0], hs[bo][1]);
    }

    // 3) h_last = outputs[T-1]
    hipMemcpyAsync(out + (size_t)TSTEPS * NH, out + (size_t)(TSTEPS - 1) * NH,
                   NH * sizeof(float), hipMemcpyDeviceToDevice, stream);
}
